// RecogNet_81905026334915
// MI455X (gfx1250) — matrix operand order repaired and checked by reading, not compiled, not run on hardware
//
#include <hip/hip_runtime.h>
#include <hip/hip_bf16.h>
#include <math.h>

// ---------------------------------------------------------------------------
// Problem constants (from the reference)
// ---------------------------------------------------------------------------
#define T_STEPS 64
#define BATCH   512
#define NB      (BATCH * 10)   // 5120 effective LSTM rows
#define HID     112
#define NGATE   (4 * HID)      // 448
#define KX      36             // x feature width
#define KREAL   (KX + HID)     // 148
#define KP      160            // K padded to multiple of 32 for WMMA
#define OUT_C   3

typedef __attribute__((ext_vector_type(16))) _Float16 v16h;
typedef __attribute__((ext_vector_type(8)))  _Float16 v8h;
typedef __attribute__((ext_vector_type(8)))  float    v8f;

__device__ __forceinline__ float sigmoidf_(float x) {
    return 1.0f / (1.0f + __expf(-x));
}

// ---------------------------------------------------------------------------
// Prep: pack combined gate weights [448][160] f16 = [W_ih | W_hh | 0pad],
//       combined bias[448] = b_ih + b_hh
// ---------------------------------------------------------------------------
__global__ void prep_weights(const float* __restrict__ w_ih,   // (448,36)
                             const float* __restrict__ w_hh,   // (448,112)
                             const float* __restrict__ b_ih,
                             const float* __restrict__ b_hh,
                             _Float16* __restrict__ Wt,        // (448,160) f16
                             float* __restrict__ bias) {       // (448)
    int idx = blockIdx.x * blockDim.x + threadIdx.x;
    if (idx >= NGATE * KP) return;
    int n = idx / KP, k = idx % KP;
    float v = 0.0f;
    if (k < KX)          v = w_ih[n * KX + k];
    else if (k < KREAL)  v = w_hh[n * HID + (k - KX)];
    Wt[idx] = (_Float16)v;
    if (k == 0) bias[n] = b_ih[n] + b_hh[n];
}

// ---------------------------------------------------------------------------
// Fused 3-stage CNN: one workgroup per (t, b) image; all intermediates in LDS.
//   conv3x3(SAME)+relu+maxpool2, three times: 1->10->20->40 channels.
//   Emits f16 features Xh[t][b*10+q][36].
// ---------------------------------------------------------------------------
__global__ void __launch_bounds__(256)
conv_stack(const float* __restrict__ imgs,                       // (T,B,1,24,24)
           const float* __restrict__ w1, const float* __restrict__ b1,
           const float* __restrict__ w2, const float* __restrict__ b2,
           const float* __restrict__ w3, const float* __restrict__ b3,
           _Float16* __restrict__ Xh) {                          // (T,NB,36)
    __shared__ float simg[24 * 24];        // 576
    __shared__ float f1[10 * 12 * 12];     // 1440
    __shared__ float f2[20 * 6 * 6];       // 720

    const int tb = blockIdx.x;
    const int t  = tb / BATCH;
    const int b  = tb % BATCH;
    const int tid = threadIdx.x;

    // ---- load image ----
    const float* img = imgs + (size_t)tb * 576;
    for (int i = tid; i < 576; i += 256) simg[i] = img[i];
    __syncthreads();

    // ---- stage 1: (1,24,24) -> conv 10ch -> pool -> f1 (10,12,12) ----
    for (int idx = tid; idx < 10 * 144; idx += 256) {
        int c  = idx / 144;
        int py = (idx % 144) / 12, px = idx % 12;
        float mx = 0.0f;  // relu => pooled max >= 0
        #pragma unroll
        for (int dy = 0; dy < 2; ++dy)
        #pragma unroll
        for (int dx = 0; dx < 2; ++dx) {
            int Y = 2 * py + dy, X = 2 * px + dx;
            float s = b1[c];
            #pragma unroll
            for (int ky = 0; ky < 3; ++ky) {
                int iy = Y - 1 + ky;
                if ((unsigned)iy >= 24u) continue;
                #pragma unroll
                for (int kx = 0; kx < 3; ++kx) {
                    int ix = X - 1 + kx;
                    if ((unsigned)ix >= 24u) continue;
                    s += simg[iy * 24 + ix] * w1[(c * 3 + ky) * 3 + kx];
                }
            }
            mx = fmaxf(mx, fmaxf(s, 0.0f));
        }
        f1[idx] = mx;
    }
    __syncthreads();

    // ---- stage 2: (10,12,12) -> conv 20ch -> pool -> f2 (20,6,6) ----
    for (int idx = tid; idx < 20 * 36; idx += 256) {
        int c  = idx / 36;
        int py = (idx % 36) / 6, px = idx % 6;
        float mx = 0.0f;
        #pragma unroll
        for (int dy = 0; dy < 2; ++dy)
        #pragma unroll
        for (int dx = 0; dx < 2; ++dx) {
            int Y = 2 * py + dy, X = 2 * px + dx;
            float s = b2[c];
            for (int ci = 0; ci < 10; ++ci) {
                const float* wp = w2 + ((c * 10 + ci) * 9);
                const float* fp = f1 + ci * 144;
                #pragma unroll
                for (int ky = 0; ky < 3; ++ky) {
                    int iy = Y - 1 + ky;
                    if ((unsigned)iy >= 12u) continue;
                    #pragma unroll
                    for (int kx = 0; kx < 3; ++kx) {
                        int ix = X - 1 + kx;
                        if ((unsigned)ix >= 12u) continue;
                        s += fp[iy * 12 + ix] * wp[ky * 3 + kx];
                    }
                }
            }
            mx = fmaxf(mx, fmaxf(s, 0.0f));
        }
        f2[idx] = mx;
    }
    __syncthreads();

    // ---- stage 3: (20,6,6) -> conv 40ch -> pool -> (40,3,3) -> Xh f16 ----
    for (int idx = tid; idx < 40 * 9; idx += 256) {
        int c  = idx / 9;
        int py = (idx % 9) / 3, px = idx % 3;
        float mx = 0.0f;
        #pragma unroll
        for (int dy = 0; dy < 2; ++dy)
        #pragma unroll
        for (int dx = 0; dx < 2; ++dx) {
            int Y = 2 * py + dy, X = 2 * px + dx;
            float s = b3[c];
            for (int ci = 0; ci < 20; ++ci) {
                const float* wp = w3 + ((c * 20 + ci) * 9);
                const float* fp = f2 + ci * 36;
                #pragma unroll
                for (int ky = 0; ky < 3; ++ky) {
                    int iy = Y - 1 + ky;
                    if ((unsigned)iy >= 6u) continue;
                    #pragma unroll
                    for (int kx = 0; kx < 3; ++kx) {
                        int ix = X - 1 + kx;
                        if ((unsigned)ix >= 6u) continue;
                        s += fp[iy * 6 + ix] * wp[ky * 3 + kx];
                    }
                }
            }
            mx = fmaxf(mx, fmaxf(s, 0.0f));
        }
        // flat (c,3,3) index -> LSTM row q = idx/36, feature k = idx%36
        int q = idx / KX, k = idx % KX;
        Xh[((size_t)t * NB + b * 10 + q) * KX + k] = (_Float16)mx;
    }
}

// ---------------------------------------------------------------------------
// Persistent LSTM scan + fused softmax head.
// Grid: NB/16 = 320 blocks, 128 threads (4 waves). The recurrence is
// block-local (row m's state depends only on row m's gates), so each block
// runs all 64 timesteps:
//   - c state lives in registers (14 f32/thread)
//   - h state lives in the LDS A-tile (f16), rewritten each step
//   - gates via v_wmma_f32_16x16x32_f16 (4 waves x 7 N-tiles x 5 K-steps)
//   - weights stay L2-resident; x_t streamed from the f16 feature buffer
// Final step stages h (f32) in LDS and computes softmax(h @ wf^T + bf).
// ---------------------------------------------------------------------------
__global__ void __launch_bounds__(128)
lstm_scan(const _Float16* __restrict__ Xh,    // (T,NB,36)
          const _Float16* __restrict__ Wt,    // (448,160)
          const float*    __restrict__ bias,  // (448)
          const float*    __restrict__ h0,    // (112)
          const float*    __restrict__ c0,    // (112)
          const float*    __restrict__ wf,    // (3,112)
          const float*    __restrict__ bf,    // (3)
          float* __restrict__ out) {          // (NB,3)
    __shared__ __align__(32) _Float16 Atile[16][KP];   // 5 KB: [x | h | 0]
    __shared__ float Gt[16][NGATE];                    // 28 KB gates
    __shared__ float Hs[16][HID];                      // 7 KB final h (f32)
    __shared__ float Zs[16][OUT_C];                    // head logits

    const int tid  = threadIdx.x;
    const int wave = tid >> 5;
    const int lane = tid & 31;
    const int m0   = blockIdx.x * 16;

    // ---- init: h region of A-tile from h0, K pad region zero ----
    for (int idx = tid; idx < 16 * KP; idx += 128) {
        int r2 = idx / KP, k = idx % KP;
        if (k >= KX) {
            float v = (k < KREAL) ? h0[k - KX] : 0.0f;
            Atile[r2][k] = (_Float16)v;
        }
    }
    // ---- c state in registers: element idx = tid + 128*u ----
    float creg[14];
    #pragma unroll
    for (int u = 0; u < 14; ++u) creg[u] = c0[(tid + 128 * u) % HID];

    const int r = lane & 15;    // A row / D column-within-group
    const int h = lane >> 4;    // lane-half selector

    for (int t = 0; t < T_STEPS; ++t) {
        // ---- stream x_t into the A-tile (f16) ----
        for (int idx = tid; idx < 16 * KX; idx += 128) {
            int r2 = idx / KX, k = idx % KX;
            Atile[r2][k] = Xh[((size_t)t * NB + m0 + r2) * KX + k];
        }
        __syncthreads();   // A-tile (x + h) complete before GEMM

        // ---- gates GEMM: 4 waves x 7 tiles of 16x16, K = 160 ----
        #pragma unroll
        for (int ti = 0; ti < 7; ++ti) {
            const int n0  = (wave * 7 + ti) * 16;
            const int col = n0 + (lane & 15);
            v8f acc = {};
            __builtin_prefetch(Wt + (size_t)((n0 + 16) & (NGATE - 1)) * KP, 0, 1);
            #pragma unroll
            for (int kk = 0; kk < KP / 32; ++kk) {
                const int k0 = kk * 32;
                // A fragment: 16 contiguous halves along K (k = 16h + s), matching B
                v8h lo = *(const v8h*)&Atile[r][k0 + 16 * h];
                v8h hi = *(const v8h*)&Atile[r][k0 + 16 * h + 8];
                v16h a;
                #pragma unroll
                for (int e = 0; e < 8; ++e) { a[e] = lo[e]; a[e + 8] = hi[e]; }
                // B fragment: 16 contiguous halves along K of weight row `col`
                v16h bfrag = *(const v16h*)(Wt + (size_t)col * KP + k0 + 16 * h);
                acc = __builtin_amdgcn_wmma_f32_16x16x32_f16(
                    false, a, false, bfrag, (short)0, acc, false, false);
            }
            // D layout: vgpr j -> row 8*h + j, column `col`
            #pragma unroll
            for (int j = 0; j < 8; ++j) Gt[8 * h + j][col] = acc[j];
        }
        __syncthreads();   // gates complete before cell update

        // ---- elementwise LSTM cell update (c in regs, h -> A-tile f16) ----
        #pragma unroll
        for (int u = 0; u < 14; ++u) {
            int idx = tid + 128 * u;
            int m = idx / HID, j = idx % HID;
            float ig = sigmoidf_(Gt[m][j          ] + bias[j          ]);
            float fg = sigmoidf_(Gt[m][HID     + j] + bias[HID     + j]);
            float g  = tanhf    (Gt[m][2 * HID + j] + bias[2 * HID + j]);
            float og = sigmoidf_(Gt[m][3 * HID + j] + bias[3 * HID + j]);
            float cn = fg * creg[u] + ig * g;
            float hn = og * tanhf(cn);
            creg[u] = cn;
            Atile[m][KX + j] = (_Float16)hn;
            if (t == T_STEPS - 1) Hs[m][j] = hn;
        }
        // no sync needed here: next iteration's x-write touches a disjoint
        // LDS region, and the top-of-loop __syncthreads orders h-writes
        // against the next GEMM's reads (and Gt writes against Gt reads).
    }
    __syncthreads();

    // ---- fused head: z = h @ wf^T + bf (48 lanes), then row softmax ----
    for (int idx = tid; idx < 16 * OUT_C; idx += 128) {
        int m = idx / OUT_C, o = idx % OUT_C;
        float s = bf[o];
        const float* wr = wf + o * HID;
        for (int j = 0; j < HID; ++j) s += Hs[m][j] * wr[j];
        Zs[m][o] = s;
    }
    __syncthreads();
    for (int m = tid; m < 16; m += 128) {
        float z0 = Zs[m][0], z1 = Zs[m][1], z2 = Zs[m][2];
        float mx = fmaxf(z0, fmaxf(z1, z2));
        float e0 = __expf(z0 - mx), e1 = __expf(z1 - mx), e2 = __expf(z2 - mx);
        float inv = 1.0f / (e0 + e1 + e2);
        size_t ro = (size_t)(m0 + m) * OUT_C;
        out[ro + 0] = e0 * inv;
        out[ro + 1] = e1 * inv;
        out[ro + 2] = e2 * inv;
    }
}

// ---------------------------------------------------------------------------
// Launch
// ---------------------------------------------------------------------------
extern "C" void kernel_launch(void* const* d_in, const int* in_sizes, int n_in,
                              void* d_out, int out_size, void* d_ws, size_t ws_size,
                              hipStream_t stream) {
    const float* imgs = (const float*)d_in[0];
    const float* w1   = (const float*)d_in[1];
    const float* b1   = (const float*)d_in[2];
    const float* w2   = (const float*)d_in[3];
    const float* b2   = (const float*)d_in[4];
    const float* w3   = (const float*)d_in[5];
    const float* b3   = (const float*)d_in[6];
    const float* w_ih = (const float*)d_in[7];
    const float* w_hh = (const float*)d_in[8];
    const float* b_ih = (const float*)d_in[9];
    const float* b_hh = (const float*)d_in[10];
    const float* wf   = (const float*)d_in[11];
    const float* bf   = (const float*)d_in[12];
    const float* h0   = (const float*)d_in[13];
    const float* c0   = (const float*)d_in[14];

    // workspace carve-up (all 256B aligned)
    char* ws = (char*)d_ws;
    size_t off = 0;
    _Float16* Wt   = (_Float16*)(ws + off); off += (size_t)NGATE * KP * sizeof(_Float16);  // 143360
    float*    bias = (float*)   (ws + off); off += 2048;                                   // 448*4 padded
    _Float16* Xh   = (_Float16*)(ws + off); off += (size_t)T_STEPS * NB * KX * sizeof(_Float16);
    (void)off; (void)ws_size; (void)in_sizes; (void)n_in; (void)out_size;

    prep_weights<<<(NGATE * KP + 255) / 256, 256, 0, stream>>>(w_ih, w_hh, b_ih, b_hh, Wt, bias);

    // CNN features for all timesteps in parallel (no recurrence in the CNN)
    conv_stack<<<T_STEPS * BATCH, 256, 0, stream>>>(imgs, w1, b1, w2, b2, w3, b3, Xh);

    // single persistent kernel: 64-step LSTM scan + fused softmax head
    lstm_scan<<<NB / 16, 128, 0, stream>>>(Xh, Wt, bias, h0, c0, wf, bf, (float*)d_out);
}
